// BiLSTM_42408507080880
// MI455X (gfx1250) — matrix operand order repaired and checked by reading, not compiled, not run on hardware
//
#include <hip/hip_runtime.h>

// ---------------------------------------------------------------------------
// BiLSTM for MI455X (gfx1250): bf16 WMMA GEMMs, f32 accumulate.
// Pass 0: convert x, Wx to bf16 in workspace (memory-bound, hw v_cvt).
// Phase 1: gx = x @ Wx^T + bx  (bf16 WMMA, 4m x 2n tiles per wave, no VALU
//          conversion in the K-loop).
// Phase 2: persistent per-direction scan; Wh slice resident in LDS (bf16),
//          h exchanged via L2-resident bf16 ping-pong buffers, cell state in
//          registers, per-direction atomic grid barrier each timestep, and
//          global_prefetch of the next step's gx slab overlapped with WMMA.
// ---------------------------------------------------------------------------

typedef __attribute__((ext_vector_type(16))) __bf16          v16bf;
typedef __attribute__((ext_vector_type(8)))  float           v8f;
typedef __attribute__((ext_vector_type(4)))  float           v4f;
typedef __attribute__((ext_vector_type(8)))  unsigned short  v8us;
typedef __attribute__((ext_vector_type(16))) unsigned short  v16us;

union BfVec { v16us u; v16bf b; };

#define Bq 32
#define Tq 512
#define Iq 512
#define Hq 512
#define Gq 2048          // 4*H
#define NWG_DIR 32       // workgroups per direction (phase 2)
#define HID_WG 16        // hidden units owned per workgroup
#define NLOC 64          // gate columns per workgroup (4*HID_WG)
#define WH_STRIDE 520    // LDS row stride in bf16 elems (512 + 8 pad)
#define GATE_STRIDE 68   // LDS gate-tile row stride in f32 (64 + 4 pad)

// hardware conversion: lowers to v_cvt bf16, not integer emulation
static __device__ __forceinline__ unsigned short f2bf(float f) {
  union { __bf16 b; unsigned short u; } v;
  v.b = (__bf16)f;
  return v.u;
}

static __device__ __forceinline__ float sigmoidf_(float x) {
  return 1.0f / (1.0f + __expf(-x));
}

// ---------------------------------------------------------------------------
// Pass 0: f32 -> bf16 bulk convert (8 elements / thread, b128 in, b128 out)
// ---------------------------------------------------------------------------
__global__ __launch_bounds__(256) void cvt_f32_bf16(const float* __restrict__ src,
                                                    unsigned short* __restrict__ dst,
                                                    int n) {
  const int base = (blockIdx.x * 256 + threadIdx.x) * 8;
  if (base >= n) return;
  v4f f0 = *(const v4f*)(src + base);
  v4f f1 = *(const v4f*)(src + base + 4);
  v8us o;
  #pragma unroll
  for (int e = 0; e < 4; ++e) { o[e] = f2bf(f0[e]); o[e + 4] = f2bf(f1[e]); }
  *(v8us*)(dst + base) = o;
}

// ---------------------------------------------------------------------------
// Phase 1: bf16 GEMM, each wave owns a 4m x 2n block of 16x16 tiles.
// gx stored [T][B][4H] so phase 2 reads a contiguous 256 KB slab per step.
// ---------------------------------------------------------------------------
__global__ __launch_bounds__(256) void gx_gemm(const unsigned short* __restrict__ xb,
                                               const unsigned short* __restrict__ wxb,
                                               const float* __restrict__ bx,
                                               float* __restrict__ gx) {
  const int lane = threadIdx.x & 31;
  const int wid  = blockIdx.x * 8 + (threadIdx.x >> 5);
  const int mg   = wid >> 6;                 // 256 m-groups (4 tiles each)
  const int ng   = wid & 63;                 // 64  n-groups (2 tiles each)
  const int l16  = (lane >= 16) ? 1 : 0;

  const unsigned short* xr[4];
  const unsigned short* wr[2];
  #pragma unroll
  for (int mi = 0; mi < 4; ++mi)
    xr[mi] = xb + (size_t)((mg * 4 + mi) * 16 + (lane & 15)) * Iq;
  #pragma unroll
  for (int ni = 0; ni < 2; ++ni)
    wr[ni] = wxb + (size_t)((ng * 2 + ni) * 16 + (lane & 15)) * Iq;

  v8f acc[4][2];
  #pragma unroll
  for (int ni = 0; ni < 2; ++ni) {
    const float bias = bx[(ng * 2 + ni) * 16 + (lane & 15)];
    #pragma unroll
    for (int mi = 0; mi < 4; ++mi)
      #pragma unroll
      for (int r = 0; r < 8; ++r) acc[mi][ni][r] = bias;
  }

  #pragma unroll 2
  for (int kc = 0; kc < Iq / 32; ++kc) {
    const int k0 = kc * 32;
    const int ha = k0 + (l16 ? 8 : 0);   // A: lanes<16 K={k0..+7,k0+16..+23}
    const int hk = k0 + (l16 ? 8 : 0);   // B: lanes<16 K={k0..+7,k0+16..+23}
    BfVec bv[2];
    #pragma unroll
    for (int ni = 0; ni < 2; ++ni) {
      v8us b0 = *(const v8us*)(wr[ni] + hk);
      v8us b1 = *(const v8us*)(wr[ni] + hk + 16);
      bv[ni].u = __builtin_shufflevector(b0, b1,
                   0,1,2,3,4,5,6,7,8,9,10,11,12,13,14,15);
    }
    #pragma unroll
    for (int mi = 0; mi < 4; ++mi) {
      v8us a0 = *(const v8us*)(xr[mi] + ha);
      v8us a1 = *(const v8us*)(xr[mi] + ha + 16);
      BfVec av;
      av.u = __builtin_shufflevector(a0, a1,
               0,1,2,3,4,5,6,7,8,9,10,11,12,13,14,15);
      #pragma unroll
      for (int ni = 0; ni < 2; ++ni)
        acc[mi][ni] = __builtin_amdgcn_wmma_f32_16x16x32_bf16(
            false, av.b, false, bv[ni].b, (short)0, acc[mi][ni], false, false);
    }
  }

  // C/D layout: VGPR r -> M=r (lanes 0-15) / r+8 (lanes 16-31), N = lane%16
  #pragma unroll
  for (int mi = 0; mi < 4; ++mi) {
    #pragma unroll
    for (int ni = 0; ni < 2; ++ni) {
      const int n0 = (ng * 2 + ni) * 16;
      #pragma unroll
      for (int r = 0; r < 8; ++r) {
        const int m = (mg * 4 + mi) * 16 + r + (l16 ? 8 : 0);
        const int t = m & (Tq - 1);
        const int b = m >> 9;
        gx[((size_t)(t * Bq + b) << 11) + n0 + (lane & 15)] = acc[mi][ni][r];
      }
    }
  }
}

// ---------------------------------------------------------------------------
// Phase 2: persistent recurrent scan. grid = 2 dirs * 32 WGs, 256 thr each.
// ---------------------------------------------------------------------------
__global__ __launch_bounds__(256) void lstm_scan(const float* __restrict__ gx,
                                                 const float* __restrict__ Wh,
                                                 const float* __restrict__ bh,
                                                 unsigned short* __restrict__ hbuf,
                                                 int* __restrict__ counters,
                                                 float* __restrict__ out) {
  extern __shared__ char smem[];
  unsigned short* whLds   = (unsigned short*)smem;                       // [64][520] bf16
  float*          gatesLds = (float*)(smem + NLOC * WH_STRIDE * 2);      // [32][68]
  float*          bhLds    = (float*)(smem + NLOC * WH_STRIDE * 2
                                           + Bq * GATE_STRIDE * 4);      // [64]

  const int tid = threadIdx.x;
  const int dir = blockIdx.x >> 5;
  const int wg  = blockIdx.x & 31;
  const int j0  = wg * HID_WG;

  // ---- one-time: stage Wh slice (bf16) + bh slice into LDS ----------------
  // local row lr = g*16 + jj  <->  global Wh row G = g*512 + j0 + jj
  for (int idx = tid; idx < NLOC * (Hq / 4); idx += 256) {
    const int lr = idx >> 7;
    const int k  = (idx & 127) * 4;
    const int G  = (lr >> 4) * Hq + j0 + (lr & 15);
    v4f w = *(const v4f*)(Wh + (size_t)G * Hq + k);
    unsigned short* d = whLds + lr * WH_STRIDE + k;
    d[0] = f2bf(w[0]); d[1] = f2bf(w[1]); d[2] = f2bf(w[2]); d[3] = f2bf(w[3]);
  }
  if (tid < NLOC) bhLds[tid] = bh[(tid >> 4) * Hq + j0 + (tid & 15)];
  __syncthreads();

  const int lane = tid & 31;
  const int wv   = tid >> 5;
  const int mt   = wv >> 2;                      // 2 m-tiles (batch 32)
  const int nt   = wv & 3;                       // 4 n-tiles (64 gate cols)
  const int l16  = (lane >= 16) ? 1 : 0;
  const int arow = mt * 16 + (lane & 15);        // batch row for A
  const int lrB  = nt * 16 + (lane & 15);        // local gate col for B
  unsigned short* hb_base = hbuf + dir * 2 * (Bq * Hq);

  // cell state in registers: elements e = tid*2 + {0,1};  b = e>>4, jj = e&15
  float cst[2] = {0.f, 0.f};
  const int eb0 = (tid * 2) >> 4;
  const int ej0 = (tid * 2) & 15;

  // prefetch step-0 gx slab (gfx1250 global_prefetch_b8)
  {
    const int t0 = dir ? (Tq - 1) : 0;
    const float* g0 = gx + (size_t)t0 * (Bq * Gq) + eb0 * Gq + j0 + ej0;
    #pragma unroll
    for (int g = 0; g < 4; ++g) __builtin_prefetch(g0 + g * Hq, 0, 1);
  }

  for (int s = 0; s < Tq; ++s) {
    const int t_in = dir ? (Tq - 1 - s) : s;
    const unsigned short* hrow = hb_base + (s & 1) * (Bq * Hq) + arow * Hq;
    const unsigned short* wr   = whLds + lrB * WH_STRIDE;

    // prefetch NEXT step's gx slab; latency hidden under the WMMA K-loop
    if (s + 1 < Tq) {
      const int t_nx = dir ? (Tq - 2 - s) : (s + 1);
      const float* gn = gx + (size_t)t_nx * (Bq * Gq) + eb0 * Gq + j0 + ej0;
      #pragma unroll
      for (int g = 0; g < 4; ++g) __builtin_prefetch(gn + g * Hq, 0, 1);
    }

    // ---- gates tile: h_{s-1} @ Wh_slice^T via bf16 WMMA -------------------
    v8f acc = {};
    #pragma unroll 4
    for (int kc = 0; kc < Hq / 32; ++kc) {
      const int k0 = kc * 32;
      const int ha = k0 + (l16 ? 8 : 0);
      v8us a0 = *(const v8us*)(hrow + ha);          // K = ha..ha+7
      v8us a1 = *(const v8us*)(hrow + ha + 16);     // K = ha+16..ha+23
      const int hk = k0 + (l16 ? 8 : 0);
      v8us b0 = *(const v8us*)(wr + hk);            // ds_load_b128
      v8us b1 = *(const v8us*)(wr + hk + 16);
      BfVec av, bv;
      av.u = __builtin_shufflevector(a0, a1, 0,1,2,3,4,5,6,7,8,9,10,11,12,13,14,15);
      bv.u = __builtin_shufflevector(b0, b1, 0,1,2,3,4,5,6,7,8,9,10,11,12,13,14,15);
      acc = __builtin_amdgcn_wmma_f32_16x16x32_bf16(false, av.b, false, bv.b,
                                                    (short)0, acc, false, false);
    }

    // ---- exchange gate tile through LDS -----------------------------------
    #pragma unroll
    for (int r = 0; r < 8; ++r) {
      const int m = mt * 16 + r + (l16 ? 8 : 0);
      gatesLds[m * GATE_STRIDE + lrB] = acc[r];
    }
    __syncthreads();

    // ---- element-wise cell update (2 cells per thread) --------------------
    const float* gxT = gx + (size_t)t_in * (Bq * Gq);
    unsigned short* hout = hb_base + ((s + 1) & 1) * (Bq * Hq);
    #pragma unroll
    for (int k2 = 0; k2 < 2; ++k2) {
      const int b  = eb0;
      const int jj = ej0 + k2;
      float pre[4];
      #pragma unroll
      for (int g = 0; g < 4; ++g) {
        const int cc = g * 16 + jj;
        pre[g] = gatesLds[b * GATE_STRIDE + cc] + bhLds[cc]
               + gxT[b * Gq + g * Hq + j0 + jj];
      }
      const float ig = sigmoidf_(pre[0]);
      const float fg = sigmoidf_(pre[1]);
      const float gg = tanhf(pre[2]);
      const float og = sigmoidf_(pre[3]);
      const float cy = cst[k2] * fg + ig * gg;
      cst[k2] = cy;
      const float hy = og * tanhf(cy);
      hout[b * Hq + j0 + jj] = f2bf(hy);
      out[((size_t)(b * Tq + s) << 10) + dir * Hq + j0 + jj] = hy;
    }

    // ---- per-direction grid barrier (monotonic counter) -------------------
    __syncthreads();
    __threadfence();                       // release h writes to device scope
    if (tid == 0) {
      atomicAdd(&counters[dir], 1);
      const int target = NWG_DIR * (s + 1);
      while (__hip_atomic_load(&counters[dir], __ATOMIC_RELAXED,
                               __HIP_MEMORY_SCOPE_AGENT) < target) {
        __builtin_amdgcn_s_sleep(1);
      }
    }
    __syncthreads();
    __threadfence();                       // acquire: invalidate stale h lines
  }
}

// ---------------------------------------------------------------------------
extern "C" void kernel_launch(void* const* d_in, const int* in_sizes, int n_in,
                              void* d_out, int out_size, void* d_ws, size_t ws_size,
                              hipStream_t stream) {
  const float* x  = (const float*)d_in[0];
  const float* Wx = (const float*)d_in[1];
  const float* bx = (const float*)d_in[2];
  const float* Wh = (const float*)d_in[3];
  const float* bh = (const float*)d_in[4];
  float* out = (float*)d_out;

  const size_t GX_BYTES  = (size_t)Tq * Bq * Gq * sizeof(float);          // 128 MB
  const size_t XBF_BYTES = (size_t)Bq * Tq * Iq * sizeof(unsigned short); // 16 MB
  const size_t WXB_BYTES = (size_t)Gq * Iq * sizeof(unsigned short);      // 2 MB
  const size_t HB_BYTES  = (size_t)4 * Bq * Hq * sizeof(unsigned short);  // 4x32KB

  char* ws = (char*)d_ws;
  float*          gx       = (float*)ws;
  unsigned short* xbf      = (unsigned short*)(ws + GX_BYTES);
  unsigned short* wxbf     = (unsigned short*)(ws + GX_BYTES + XBF_BYTES);
  unsigned short* hbuf     = (unsigned short*)(ws + GX_BYTES + XBF_BYTES + WXB_BYTES);
  int*            counters = (int*)(ws + GX_BYTES + XBF_BYTES + WXB_BYTES + HB_BYTES);

  // zero ping-pong h buffers (h0 = 0) and barrier counters every call
  hipMemsetAsync(hbuf, 0, HB_BYTES + 256, stream);

  // Pass 0: bf16 conversion of x (8.4M elems) and Wx (1M elems)
  const int nx = Bq * Tq * Iq;
  const int nw = Gq * Iq;
  cvt_f32_bf16<<<nx / (8 * 256), 256, 0, stream>>>(x, xbf, nx);
  cvt_f32_bf16<<<nw / (8 * 256), 256, 0, stream>>>(Wx, wxbf, nw);

  // Phase 1: 1024x128 tiles, 4m x 2n per wave -> 16384 waves / 8 per block
  gx_gemm<<<2048, 256, 0, stream>>>(xbf, wxbf, bx, gx);

  // Phase 2: persistent scan, 2 dirs * 32 WGs
  const int smem = NLOC * WH_STRIDE * 2 + Bq * GATE_STRIDE * 4 + NLOC * 4; // ~75.5 KB
  hipFuncSetAttribute((const void*)lstm_scan,
                      hipFuncAttributeMaxDynamicSharedMemorySize, smem);
  lstm_scan<<<2 * NWG_DIR, 256, smem, stream>>>(gx, Wh, bh, hbuf, counters, out);
}
